// CyberMoE_64424509440620
// MI455X (gfx1250) — matrix operand order repaired and checked by reading, not compiled, not run on hardware
//
#include <hip/hip_runtime.h>
#include <hip/hip_bf16.h>
#include <math.h>

#define BDIM 32
#define SDIM 512
#define HDIM 768
#define EDIM 5
#define LDIM 2

typedef unsigned int u32;
typedef unsigned short u16;

typedef __bf16 bf16t;
typedef bf16t v16bf __attribute__((ext_vector_type(16)));
typedef float v8f __attribute__((ext_vector_type(8)));

union Frag {
    v16bf v;
    u32 u[8];
    uint4 q[2];
};

__device__ __forceinline__ u16 f2bf(float f) {
    u32 x = __float_as_uint(f);
    u32 r = x + 0x7FFFu + ((x >> 16) & 1u);
    return (u16)(r >> 16);
}
__device__ __forceinline__ float bf2f(u16 s) {
    return __uint_as_float(((u32)s) << 16);
}
__device__ __forceinline__ float gelu_erf(float x) {
    return 0.5f * x * (1.0f + erff(x * 0.70710678118654752f));
}

#define LDSTRIDE 40  // halves per row (80 B): keeps b128 alignment, spreads banks

// ---------------------------------------------------------------------------
// Generic bf16 WMMA GEMM: C[M,N] = scale * (A[M,K] @ B[K,N]) (+bias[n]) (+C)
// Block tile 128x128, 256 threads (8 waves, wave32), K-chunk 32,
// v_wmma_f32_16x16x32_bf16, double-buffered LDS, software-pipelined loads.
// LDS layouts: lA[m][k] stride 40, lB[n][k] stride 40 -> all fragment loads
// are 2 x ds_load_b128 (no scalar gathers).
// grid = (N/128, M/128, batch). TRANSB: B[k,n] = Bsrc[n*ldb + k].
// ---------------------------------------------------------------------------
template<bool TRANSB, bool ACCUM, bool OUTBF, bool HASBIAS>
__global__ __launch_bounds__(256)
void gemm_bf16_wmma(const u16* __restrict__ A, int lda, size_t sA,
                    const u16* __restrict__ Bm, int ldb, size_t sB,
                    const float* __restrict__ bias, size_t sBias,
                    void* __restrict__ C, int ldc, size_t sC,
                    int K, float scale)
{
    __shared__ u16 lA[2][128 * LDSTRIDE];
    __shared__ u16 lB[2][128 * LDSTRIDE];
    const int t = threadIdx.x;
    const int lane = t & 31;
    const int w = t >> 5;
    const int z = blockIdx.z;
    const int m0 = blockIdx.y * 128;
    const int n0 = blockIdx.x * 128;
    const u16* Ab = A + (size_t)z * sA;
    const u16* Bb = Bm + (size_t)z * sB;

    const int rowBase = (w & 3) * 32;   // 4 waves down
    const int colBase = (w >> 2) * 64;  // 2 waves across
    const int nl = lane & 15;
    const int khA = ((lane >> 4) & 1) * 8;  // A frag K-half select
    const int kbB = (lane >> 4) * 8;        // B frag K-half select

    v8f acc[2][4];
    #pragma unroll
    for (int ti = 0; ti < 2; ++ti)
        #pragma unroll
        for (int tj = 0; tj < 4; ++tj)
            #pragma unroll
            for (int v = 0; v < 8; ++v) acc[ti][tj][v] = 0.0f;

    // loader decompositions
    const int ar = t >> 1, aoff = (t & 1) * 16;   // A: 2 thr/row, 16 halves each
    const int bp = t & 15, bnc = (t >> 4) * 8;    // B row-major: k-pair 2*bp, n-chunk
    const int tn = t >> 1, tkoff = (t & 1) * 16;  // B transposed: contiguous copy

    uint4 fa0, fa1, fb0, fb1;  // in-flight tile registers

    auto fetch = [&](int kt) {
        {
            const uint4* p = (const uint4*)(Ab + (size_t)(m0 + ar) * lda + kt + aoff);
            fa0 = p[0];
            fa1 = p[1];
        }
        if (TRANSB) {
            const uint4* p = (const uint4*)(Bb + (size_t)(n0 + tn) * ldb + kt + tkoff);
            fb0 = p[0];
            fb1 = p[1];
        } else {
            const int row = kt + 2 * bp;
            fb0 = *(const uint4*)(Bb + (size_t)row * ldb + n0 + bnc);
            fb1 = *(const uint4*)(Bb + (size_t)(row + 1) * ldb + n0 + bnc);
        }
    };
    auto stor = [&](int buf) {
        *(uint4*)&lA[buf][ar * LDSTRIDE + aoff] = fa0;
        *(uint4*)&lA[buf][ar * LDSTRIDE + aoff + 8] = fa1;
        if (TRANSB) {
            *(uint4*)&lB[buf][tn * LDSTRIDE + tkoff] = fb0;
            *(uint4*)&lB[buf][tn * LDSTRIDE + tkoff + 8] = fb1;
        } else {
            const u32 r0[4] = {fb0.x, fb0.y, fb0.z, fb0.w};
            const u32 r1[4] = {fb1.x, fb1.y, fb1.z, fb1.w};
            #pragma unroll
            for (int j = 0; j < 8; ++j) {
                u32 lo = (r0[j >> 1] >> ((j & 1) * 16)) & 0xffffu;
                u32 hi = (r1[j >> 1] >> ((j & 1) * 16)) & 0xffffu;
                *(u32*)&lB[buf][(bnc + j) * LDSTRIDE + 2 * bp] = lo | (hi << 16);
            }
        }
    };
    auto compute = [&](int buf) {
        Frag af[2];
        #pragma unroll
        for (int ti = 0; ti < 2; ++ti) {
            const u16* ap = &lA[buf][(rowBase + ti * 16 + nl) * LDSTRIDE];
            af[ti].q[0] = *(const uint4*)&ap[khA];       // K = khA .. khA+7
            af[ti].q[1] = *(const uint4*)&ap[16 + khA];  // K = 16+khA .. 16+khA+7
        }
        Frag bfr[4];
        #pragma unroll
        for (int tj = 0; tj < 4; ++tj) {
            const u16* bpp = &lB[buf][(colBase + tj * 16 + nl) * LDSTRIDE + kbB];
            bfr[tj].q[0] = *(const uint4*)&bpp[0];  // K = kbB .. kbB+7
            bfr[tj].q[1] = *(const uint4*)&bpp[16]; // K = 16+kbB .. 16+kbB+7
        }
        #pragma unroll
        for (int ti = 0; ti < 2; ++ti)
            #pragma unroll
            for (int tj = 0; tj < 4; ++tj)
                acc[ti][tj] = __builtin_amdgcn_wmma_f32_16x16x32_bf16(
                    false, af[ti].v, false, bfr[tj].v, (short)0, acc[ti][tj],
                    false, false);
    };

    const int nk = K / 32;
    fetch(0);
    stor(0);
    int cur = 0;
    for (int i = 0; i < nk; ++i) {
        __syncthreads();
        if (i + 1 < nk) fetch((i + 1) * 32);
        if (i + 2 < nk)  // stream-prefetch A (the big activation matrix)
            __builtin_prefetch(Ab + (size_t)(m0 + ar) * lda + (i + 2) * 32 + aoff, 0, 1);
        compute(cur);
        if (i + 1 < nk) stor(cur ^ 1);
        cur ^= 1;
    }

    // Epilogue: C layout VGPR v -> M = v + 8*lane[4], N = lane[3:0]
    const float* biasb = HASBIAS ? (bias + (size_t)z * sBias) : nullptr;
    const int mh = (lane >> 4) * 8;
    #pragma unroll
    for (int ti = 0; ti < 2; ++ti) {
        #pragma unroll
        for (int tj = 0; tj < 4; ++tj) {
            const int gn = n0 + colBase + tj * 16 + nl;
            #pragma unroll
            for (int v = 0; v < 8; ++v) {
                const int gm = m0 + rowBase + ti * 16 + v + mh;
                float val = acc[ti][tj][v] * scale;
                if (HASBIAS) val += biasb[gn];
                size_t ci = (size_t)gm * ldc + gn;
                if (OUTBF) {
                    u16* Cb = (u16*)C + (size_t)z * sC;
                    Cb[ci] = f2bf(val);
                } else {
                    float* Cb = (float*)C + (size_t)z * sC;
                    if (ACCUM) val += Cb[ci];
                    Cb[ci] = val;
                }
            }
        }
    }
}

// ---------------------------------------------------------------------------
__global__ void convert_kernel(const float* __restrict__ s, u16* __restrict__ d, int n) {
    int i = blockIdx.x * 256 + threadIdx.x;
    if (i < n) d[i] = f2bf(s[i]);
}

// Weff[b,i,o] = sum_d dw[b,d] * Wd[d,i,o]  -> bf16
__global__ __launch_bounds__(256)
void weff_kernel(const float* __restrict__ dw, const float* __restrict__ Wd,
                 u16* __restrict__ weff) {
    const size_t HH = (size_t)HDIM * HDIM;
    const size_t idx = (size_t)blockIdx.x * 256 + threadIdx.x;
    const int b = (int)(idx / HH);
    const size_t r = idx % HH;
    float acc = 0.0f;
    #pragma unroll
    for (int d = 0; d < EDIM; ++d)
        acc += dw[b * EDIM + d] * Wd[(size_t)d * HH + r];
    weff[idx] = f2bf(acc);
}

// beff[b,o] = sum_d dw[b,d] * bd[d,o]
__global__ void beff_kernel(const float* __restrict__ dw, const float* __restrict__ bd,
                            float* __restrict__ beff) {
    int idx = blockIdx.x * 256 + threadIdx.x;
    int b = idx / HDIM, o = idx % HDIM;
    float acc = 0.0f;
    #pragma unroll
    for (int d = 0; d < EDIM; ++d) acc += dw[b * EDIM + d] * bd[d * HDIM + o];
    beff[idx] = acc;
}

// Row-wise LayerNorm + GELU over H=768; 256 threads, 3 cols/thread.
__global__ __launch_bounds__(256)
void ln_gelu_kernel(const float* __restrict__ fused, const float* __restrict__ g,
                    const float* __restrict__ bb, u16* __restrict__ h) {
    __shared__ float red[256];
    __shared__ float s_mu, s_rs;
    const size_t row = blockIdx.x;
    const int t = threadIdx.x;
    const float* fr = fused + row * HDIM;
    float x0 = fr[t], x1 = fr[t + 256], x2 = fr[t + 512];
    red[t] = x0 + x1 + x2;
    __syncthreads();
    for (int s = 128; s > 0; s >>= 1) { if (t < s) red[t] += red[t + s]; __syncthreads(); }
    if (t == 0) s_mu = red[0] * (1.0f / HDIM);
    __syncthreads();
    float mu = s_mu;
    float d0 = x0 - mu, d1 = x1 - mu, d2 = x2 - mu;
    red[t] = d0 * d0 + d1 * d1 + d2 * d2;
    __syncthreads();
    for (int s = 128; s > 0; s >>= 1) { if (t < s) red[t] += red[t + s]; __syncthreads(); }
    if (t == 0) s_rs = rsqrtf(red[0] * (1.0f / HDIM) + 1e-5f);
    __syncthreads();
    float rs = s_rs;
    u16* hr = h + row * HDIM;
    hr[t]       = f2bf(gelu_erf(d0 * rs * g[t]       + bb[t]));
    hr[t + 256] = f2bf(gelu_erf(d1 * rs * g[t + 256] + bb[t + 256]));
    hr[t + 512] = f2bf(gelu_erf(d2 * rs * g[t + 512] + bb[t + 512]));
}

// Softmax over 512 cols, one row per block, 2 cols/thread.
__global__ __launch_bounds__(256)
void softmax_kernel(const float* __restrict__ sc, u16* __restrict__ pr) {
    __shared__ float red[256];
    __shared__ float s_mx, s_sum;
    const size_t row = blockIdx.x;
    const int t = threadIdx.x;
    const float* r = sc + row * SDIM;
    float a = r[t], b = r[t + 256];
    red[t] = fmaxf(a, b);
    __syncthreads();
    for (int s = 128; s > 0; s >>= 1) { if (t < s) red[t] = fmaxf(red[t], red[t + s]); __syncthreads(); }
    if (t == 0) s_mx = red[0];
    __syncthreads();
    float e0 = expf(a - s_mx), e1 = expf(b - s_mx);
    red[t] = e0 + e1;
    __syncthreads();
    for (int s = 128; s > 0; s >>= 1) { if (t < s) red[t] += red[t + s]; __syncthreads(); }
    if (t == 0) s_sum = red[0];
    __syncthreads();
    float inv = 1.0f / s_sum;
    u16* o = pr + row * SDIM;
    o[t] = f2bf(e0 * inv);
    o[t + 256] = f2bf(e1 * inv);
}

// dst[b,o] = invS * sum_s src[b,s,o]   (src bf16 [B,S,H]); grid(H/256, B)
__global__ __launch_bounds__(256)
void colmean_kernel(const u16* __restrict__ src, float* __restrict__ dst, float invS) {
    const int o = blockIdx.x * 256 + threadIdx.x;
    const int b = blockIdx.y;
    const u16* p = src + (size_t)b * SDIM * HDIM + o;
    float s = 0.0f;
    for (int i = 0; i < SDIM; ++i) s += bf2f(p[(size_t)i * HDIM]);
    dst[(size_t)b * HDIM + o] = s * invS;
}

// Gating head + experts + classifier; one block per batch element.
__global__ __launch_bounds__(256)
void head_kernel(const float* __restrict__ seqr, const float* __restrict__ pooled,
                 const float* __restrict__ Wea, const float* __restrict__ bea,
                 const float* __restrict__ Wg1, const float* __restrict__ bg1,
                 const float* __restrict__ Wg2, const float* __restrict__ bg2,
                 const float* __restrict__ We1, const float* __restrict__ be1,
                 const float* __restrict__ We2, const float* __restrict__ be2,
                 const float* __restrict__ Wdc, const float* __restrict__ bdc,
                 float* __restrict__ out) {
    __shared__ float s_seq[HDIM], s_pool[HDIM], s_g1[HDIM];
    __shared__ float s_h1[EDIM * HDIM];
    __shared__ float s_logit[EDIM], s_probs[EDIM], s_mask[EDIM], s_eout[EDIM * LDIM];
    const int b = blockIdx.x, t = threadIdx.x;
    for (int i = t; i < HDIM; i += 256) {
        s_seq[i]  = seqr[b * HDIM + i];
        s_pool[i] = pooled[b * HDIM + i];
    }
    __syncthreads();
    for (int o = t; o < HDIM; o += 256) {
        float acc = bg1[o];
        for (int i = 0; i < HDIM; ++i) acc += s_seq[i] * Wg1[i * HDIM + o];
        s_g1[o] = gelu_erf(acc);
    }
    __syncthreads();
    if (t < EDIM) {
        float acc = bea[t] + bg2[t];
        for (int i = 0; i < HDIM; ++i)
            acc += s_seq[i] * Wea[i * EDIM + t] + s_g1[i] * Wg2[i * EDIM + t];
        s_logit[t] = acc;
        float dl = bdc[t];
        for (int i = 0; i < HDIM; ++i) dl += s_pool[i] * Wdc[i * EDIM + t];
        out[544 + b * EDIM + t] = dl;  // domain_logits
    }
    for (int idx = t; idx < EDIM * HDIM; idx += 256) {
        int e = idx / HDIM, o = idx % HDIM;
        float acc = be1[e * HDIM + o];
        const float* W = We1 + (size_t)e * HDIM * HDIM;
        for (int i = 0; i < HDIM; ++i) acc += s_pool[i] * W[i * HDIM + o];
        s_h1[idx] = gelu_erf(acc);
    }
    __syncthreads();
    if (t == 0) {
        float mx = s_logit[0];
        for (int e = 1; e < EDIM; ++e) mx = fmaxf(mx, s_logit[e]);
        float sm = 0.0f;
        for (int e = 0; e < EDIM; ++e) { float p = expf(s_logit[e] - mx); s_probs[e] = p; sm += p; }
        for (int e = 0; e < EDIM; ++e) s_probs[e] /= sm;
        int i1 = 0;
        for (int e = 1; e < EDIM; ++e) if (s_probs[e] > s_probs[i1]) i1 = e;
        int i2 = -1;
        for (int e = 0; e < EDIM; ++e)
            if (e != i1 && (i2 < 0 || s_probs[e] > s_probs[i2])) i2 = e;
        for (int e = 0; e < EDIM; ++e) s_mask[e] = (e == i1 || e == i2) ? 1.0f : 0.0f;
    }
    __syncthreads();
    if (t < EDIM * LDIM) {
        int e = t >> 1, l = t & 1;
        float acc = be2[e * LDIM + l];
        for (int i = 0; i < HDIM; ++i)
            acc += s_h1[e * HDIM + i] * We2[(size_t)e * HDIM * LDIM + i * LDIM + l];
        acc *= s_mask[e];
        s_eout[t] = acc;
        out[224 + b * EDIM * LDIM + t] = acc;  // expert_outputs (masked)
    }
    __syncthreads();
    if (t < EDIM) out[64 + b * EDIM + t] = s_probs[t];  // routing_probs
    if (t < LDIM) {
        float f = 0.0f;
        for (int e = 0; e < EDIM; ++e) f += s_eout[e * LDIM + t] * s_probs[e];
        out[b * LDIM + t] = f;  // final_output
    }
}

// ---------------------------------------------------------------------------
extern "C" void kernel_launch(void* const* d_in, const int* in_sizes, int n_in,
                              void* d_out, int out_size, void* d_ws, size_t ws_size,
                              hipStream_t stream) {
    (void)in_sizes; (void)n_in; (void)out_size; (void)ws_size;
    const float* x   = (const float*)d_in[0];
    const float* dw  = (const float*)d_in[1];
    const float* Wd  = (const float*)d_in[2];
    const float* bd  = (const float*)d_in[3];
    const float* Wf  = (const float*)d_in[4];
    const float* bfv = (const float*)d_in[5];
    const float* lng = (const float*)d_in[6];
    const float* lnb = (const float*)d_in[7];
    const float* Wq  = (const float*)d_in[8];
    const float* bq  = (const float*)d_in[9];
    const float* Wk  = (const float*)d_in[10];
    const float* bk  = (const float*)d_in[11];
    const float* Wv  = (const float*)d_in[12];
    const float* bv  = (const float*)d_in[13];
    const float* Wea = (const float*)d_in[14];
    const float* bea = (const float*)d_in[15];
    const float* Wg1 = (const float*)d_in[16];
    const float* bg1 = (const float*)d_in[17];
    const float* Wg2 = (const float*)d_in[18];
    const float* bg2 = (const float*)d_in[19];
    const float* We1 = (const float*)d_in[20];
    const float* be1 = (const float*)d_in[21];
    const float* We2 = (const float*)d_in[22];
    const float* be2 = (const float*)d_in[23];
    const float* Wdc = (const float*)d_in[24];
    const float* bdc = (const float*)d_in[25];
    float* out = (float*)d_out;

    char* ws = (char*)d_ws;
    size_t off = 0;
    auto alloc = [&](size_t bytes) {
        size_t o = off; off = (off + bytes + 255) & ~(size_t)255; return o;
    };
    const size_t nBSH = (size_t)BDIM * SDIM * HDIM;
    const size_t HH = (size_t)HDIM * HDIM;
    u16*   xbf    = (u16*)  (ws + alloc(nBSH * 2));
    u16*   weff   = (u16*)  (ws + alloc((size_t)BDIM * HH * 2)); // later reused as ctx (smaller)
    float* beff   = (float*)(ws + alloc((size_t)BDIM * HDIM * 4));
    u16*   dabf   = (u16*)  (ws + alloc(nBSH * 2));              // later reused as probs (smaller)
    u16*   wfbf   = (u16*)  (ws + alloc(2 * HH * 2));
    float* fused  = (float*)(ws + alloc(nBSH * 4));              // later reused as scores (smaller)
    u16*   hbf    = (u16*)  (ws + alloc(nBSH * 2));
    u16*   wqbf   = (u16*)  (ws + alloc(HH * 2));
    u16*   wkbf   = (u16*)  (ws + alloc(HH * 2));
    u16*   wvbf   = (u16*)  (ws + alloc(HH * 2));
    u16*   Qbf    = (u16*)  (ws + alloc(nBSH * 2));
    u16*   Kbf    = (u16*)  (ws + alloc(nBSH * 2));
    u16*   Vbf    = (u16*)  (ws + alloc(nBSH * 2));
    float* pooled = (float*)(ws + alloc((size_t)BDIM * HDIM * 4));
    float* seqr   = (float*)(ws + alloc((size_t)BDIM * HDIM * 4));
    u16*   ctxbf  = weff;   // Weff dead after the da GEMM
    u16*   probs  = dabf;   // da dead after fusion pass 2
    float* scores = fused;  // fused dead after LN+GELU

    auto cv = [&](const float* s, u16* d, size_t n) {
        convert_kernel<<<(unsigned)((n + 255) / 256), 256, 0, stream>>>(s, d, (int)n);
    };
    cv(x,  xbf,  nBSH);
    cv(Wf, wfbf, 2 * HH);
    cv(Wq, wqbf, HH);
    cv(Wk, wkbf, HH);
    cv(Wv, wvbf, HH);

    weff_kernel<<<(unsigned)((size_t)BDIM * HH / 256), 256, 0, stream>>>(dw, Wd, weff);
    beff_kernel<<<BDIM * HDIM / 256, 256, 0, stream>>>(dw, bd, beff);

    // domain_adapted[b] = x[b] @ Weff[b] + beff[b]   (bf16 out)
    gemm_bf16_wmma<false, false, true, true>
        <<<dim3(HDIM / 128, SDIM / 128, BDIM), 256, 0, stream>>>(
            xbf, HDIM, (size_t)SDIM * HDIM, weff, HDIM, HH,
            beff, (size_t)HDIM, dabf, HDIM, (size_t)SDIM * HDIM, HDIM, 1.0f);

    // fused = x @ Wf[0:768] + bf ; fused += da @ Wf[768:1536]   (f32)
    gemm_bf16_wmma<false, false, false, true>
        <<<dim3(HDIM / 128, BDIM * SDIM / 128, 1), 256, 0, stream>>>(
            xbf, HDIM, 0, wfbf, HDIM, 0, bfv, 0, fused, HDIM, 0, HDIM, 1.0f);
    gemm_bf16_wmma<false, true, false, false>
        <<<dim3(HDIM / 128, BDIM * SDIM / 128, 1), 256, 0, stream>>>(
            dabf, HDIM, 0, wfbf + HH, HDIM, 0, nullptr, 0, fused, HDIM, 0, HDIM, 1.0f);

    ln_gelu_kernel<<<BDIM * SDIM, 256, 0, stream>>>(fused, lng, lnb, hbf);

    // Q, K, V = h @ W + b   (bf16 out, M = B*S, batch 1)
    gemm_bf16_wmma<false, false, true, true>
        <<<dim3(HDIM / 128, BDIM * SDIM / 128, 1), 256, 0, stream>>>(
            hbf, HDIM, 0, wqbf, HDIM, 0, bq, 0, Qbf, HDIM, 0, HDIM, 1.0f);
    gemm_bf16_wmma<false, false, true, true>
        <<<dim3(HDIM / 128, BDIM * SDIM / 128, 1), 256, 0, stream>>>(
            hbf, HDIM, 0, wkbf, HDIM, 0, bk, 0, Kbf, HDIM, 0, HDIM, 1.0f);
    gemm_bf16_wmma<false, false, true, true>
        <<<dim3(HDIM / 128, BDIM * SDIM / 128, 1), 256, 0, stream>>>(
            hbf, HDIM, 0, wvbf, HDIM, 0, bv, 0, Vbf, HDIM, 0, HDIM, 1.0f);

    // scores[b] = (Q[b] @ K[b]^T) / sqrt(H)   (f32 out, transposed B)
    const float inv_sqrt_h = 0.03608439182435161f;  // 1/sqrt(768)
    gemm_bf16_wmma<true, false, false, false>
        <<<dim3(SDIM / 128, SDIM / 128, BDIM), 256, 0, stream>>>(
            Qbf, HDIM, (size_t)SDIM * HDIM, Kbf, HDIM, (size_t)SDIM * HDIM,
            nullptr, 0, scores, SDIM, (size_t)SDIM * SDIM, HDIM, inv_sqrt_h);

    softmax_kernel<<<BDIM * SDIM, 256, 0, stream>>>(scores, probs);

    // ctx[b] = probs[b] @ V[b]   (bf16 out)
    gemm_bf16_wmma<false, false, true, false>
        <<<dim3(HDIM / 128, SDIM / 128, BDIM), 256, 0, stream>>>(
            probs, SDIM, (size_t)SDIM * SDIM, Vbf, HDIM, (size_t)SDIM * HDIM,
            nullptr, 0, ctxbf, HDIM, (size_t)SDIM * HDIM, SDIM, 1.0f);

    colmean_kernel<<<dim3(HDIM / 256, BDIM), 256, 0, stream>>>(hbf, pooled, 1.0f / SDIM);
    colmean_kernel<<<dim3(HDIM / 256, BDIM), 256, 0, stream>>>(ctxbf, seqr, 1.0f / SDIM);

    head_kernel<<<BDIM, 256, 0, stream>>>(seqr, pooled, Wea, bea, Wg1, bg1, Wg2, bg2,
                                          We1, be1, We2, be2, Wdc, bdc, out);
}
